// AgentAttention_3728031613533
// MI455X (gfx1250) — matrix operand order repaired and checked by reading, not compiled, not run on hardware
//
#include <hip/hip_runtime.h>
#include <hip/hip_bf16.h>
#include <math.h>

// ---------------- WMMA types / helpers ----------------
typedef float v2f __attribute__((ext_vector_type(2)));
typedef float v8f __attribute__((ext_vector_type(8)));
typedef __bf16 v16bf __attribute__((ext_vector_type(16)));
typedef unsigned int v8u __attribute__((ext_vector_type(8)));

union ABfrag { v8u u; v16bf b; };

__device__ __forceinline__ v8f wmma4(v2f a, v2f b, v8f c) {
    // D = A(16x4 f32) * B(4x16 f32) + C(16x16 f32)
    return __builtin_amdgcn_wmma_f32_16x16x4_f32(
        false, a, false, b, (short)0, c, false, false);
}

__device__ __forceinline__ v8f wmma_bf16(v16bf a, v16bf b, v8f c) {
    // D = A(16x32 bf16) * B(32x16 bf16) + C(16x16 f32)
    return __builtin_amdgcn_wmma_f32_16x16x32_bf16(
        false, a, false, b, (short)0, c, false, false);
}

__device__ __forceinline__ void zero8(v8f& x) {
#pragma unroll
    for (int e = 0; e < 8; ++e) x[e] = 0.0f;
}

__device__ __forceinline__ unsigned int pack2(__bf16 a, __bf16 b) {
    unsigned short ua = __builtin_bit_cast(unsigned short, a);
    unsigned short ub = __builtin_bit_cast(unsigned short, b);
    return (unsigned int)ua | ((unsigned int)ub << 16);
}

// Problem constants
#define BB 4
#define LL 4096
#define CC 1024
#define HH 16
#define HD 64
#define AA 49
#define NSPLIT 8       // split-K factor for agent_v
#define SCALE 0.125f   // 64^-0.5
#define SP 68          // padded LDS row stride (f32 kernels)

// ---------------- Kernel 1: v_proj = v @ W_v + b_v ----------------
// Split-bf16 GEMM: x = hi + lo (bf16 each); D += Ahi*Bhi + Ahi*Blo + Alo*Bhi,
// f32 accumulate via V_WMMA_F32_16X16X32_BF16.
// M = 16384, N = 1024, K = 1024. Tile 128x128, 8 waves, K-chunk 32.
__global__ __launch_bounds__(256) void k_vproj(const float* __restrict__ V,
                                               const float* __restrict__ W,
                                               const float* __restrict__ bias,
                                               float* __restrict__ vproj) {
    __shared__ unsigned int Ahi[128 * 17];  // [m][kp] packed bf16 pair, kp = k/2
    __shared__ unsigned int Alo[128 * 17];
    __shared__ unsigned int Bhi[16 * 128];  // [kp][n]
    __shared__ unsigned int Blo[16 * 128];
    const int tid  = threadIdx.x;
    const int lane = tid & 31, wave = tid >> 5;
    const int lane15 = lane & 15;
    const int m0 = blockIdx.x * 128;
    const int n0 = blockIdx.y * 128;

    v8f acc[8];
#pragma unroll
    for (int t = 0; t < 8; ++t) zero8(acc[t]);

    const int nn   = wave * 16 + lane15;       // column within 128-wide tile
    const int hi4  = (lane >> 4) << 2;         // A kp offset is 2*hi4 (lanes 16-31: K 16-31)
    const int bkp0 = (lane >> 4) << 3;         // B kp offset (lanes 16-31: K 16-31)

    for (int k0 = 0; k0 < CC; k0 += 32) {
        // load + split A tile (128 rows x 32 k) -> packed bf16 pairs
        for (int idx = tid; idx < 128 * 16; idx += 256) {
            int r = idx >> 4, kp = idx & 15;
            const float* p = V + (size_t)(m0 + r) * CC + k0 + (kp << 1);
            float f0 = p[0], f1 = p[1];
            __bf16 h0 = (__bf16)f0, h1 = (__bf16)f1;
            float l0 = f0 - (float)h0, l1 = f1 - (float)h1;
            Ahi[r * 17 + kp] = pack2(h0, h1);
            Alo[r * 17 + kp] = pack2((__bf16)l0, (__bf16)l1);
        }
        // load + split B tile (32 k x 128 n), packed along k
        for (int idx = tid; idx < 16 * 128; idx += 256) {
            int kp = idx >> 7, n = idx & 127;
            float f0 = W[(size_t)(k0 + (kp << 1)) * CC + n0 + n];
            float f1 = W[(size_t)(k0 + (kp << 1) + 1) * CC + n0 + n];
            __bf16 h0 = (__bf16)f0, h1 = (__bf16)f1;
            float l0 = f0 - (float)h0, l1 = f1 - (float)h1;
            Bhi[kp * 128 + n] = pack2(h0, h1);
            Blo[kp * 128 + n] = pack2((__bf16)l0, (__bf16)l1);
        }
        __syncthreads();

        ABfrag bh, bl;
#pragma unroll
        for (int v = 0; v < 8; ++v) {  // B: kp = 8*hiHalf + v
            int kp = bkp0 + v;
            bh.u[v] = Bhi[kp * 128 + nn];
            bl.u[v] = Blo[kp * 128 + nn];
        }
#pragma unroll
        for (int t = 0; t < 8; ++t) {
            int mm = t * 16 + lane15;
            ABfrag ah, al;
#pragma unroll
            for (int v = 0; v < 8; ++v) {  // A: kp = 8*hiHalf + v
                int kp = (hi4 << 1) + v;
                ah.u[v] = Ahi[mm * 17 + kp];
                al.u[v] = Alo[mm * 17 + kp];
            }
            acc[t] = wmma_bf16(ah.b, bh.b, acc[t]);
            acc[t] = wmma_bf16(ah.b, bl.b, acc[t]);
            acc[t] = wmma_bf16(al.b, bh.b, acc[t]);
        }
        __syncthreads();
    }
    const int gn = n0 + nn;
    const float bv = bias[gn];
    const int hi = (lane >> 4) << 3;
#pragma unroll
    for (int t = 0; t < 8; ++t)
#pragma unroll
        for (int r = 0; r < 8; ++r) {
            int mm = m0 + t * 16 + hi + r;
            vproj[(size_t)mm * CC + gn] = acc[t][r] + bv;
        }
}

// ---------------- Kernel 2: adaptive avg pool q -> agent (B,49,C) ----------------
__global__ __launch_bounds__(256) void k_pool(const float* __restrict__ Q,
                                              float* __restrict__ agent) {
    int idx = blockIdx.x * 256 + threadIdx.x;
    if (idx >= BB * AA * CC) return;
    int c = idx & (CC - 1);
    int t = idx >> 10;
    int i = t % AA, b = t / AA;
    int s = (i * LL) / AA;
    int e = ((i + 1) * LL + AA - 1) / AA;
    float sum = 0.f;
    const float* p = Q + ((size_t)b * LL + s) * CC + c;
    for (int l = s; l < e; ++l, p += CC) sum += *p;
    agent[idx] = sum / (float)(e - s);
}

// ---------------- Kernel 3: s1 = scale*ag @ kh^T + an_bias ----------------
// grid(L/64, B*H). M = 64 (49 agents padded), N = 64 keys, K = hd = 64.
__global__ __launch_bounds__(128) void k_s1(const float* __restrict__ K,
                                            const float* __restrict__ agent,
                                            const float* __restrict__ an_bias,
                                            float* __restrict__ attn) {
    __shared__ float Ag[64 * SP];  // [i][d]  (scaled, zero-padded rows)
    __shared__ float Ks[64 * SP];  // [l'][d]
    const int tid = threadIdx.x;
    const int lane = tid & 31, wave = tid >> 5;
    const int lane15 = lane & 15;
    const int b = blockIdx.y >> 4, h = blockIdx.y & 15;
    const int l0 = blockIdx.x * 64;

    for (int idx = tid; idx < 64 * 64; idx += 128) {
        int r = idx >> 6, c = idx & 63;
        Ag[r * SP + c] = (r < AA)
            ? SCALE * agent[(((size_t)b * AA + r) << 10) + (h << 6) + c] : 0.f;
        Ks[r * SP + c] = K[((size_t)b * LL + l0 + r) * CC + (h << 6) + c];
    }
    __syncthreads();

    v8f acc[4];
#pragma unroll
    for (int t = 0; t < 4; ++t) zero8(acc[t]);
    const int khalf = (lane >> 4) << 1;
    const int nn = wave * 16 + lane15;
    for (int ks = 0; ks < 64; ks += 4) {
        int kk = ks + khalf;
        v2f bf; bf.x = Ks[nn * SP + kk]; bf.y = Ks[nn * SP + kk + 1];
#pragma unroll
        for (int t = 0; t < 4; ++t) {
            int mm = t * 16 + lane15;
            v2f a; a.x = Ag[mm * SP + kk]; a.y = Ag[mm * SP + kk + 1];
            acc[t] = wmma4(a, bf, acc[t]);
        }
    }
    const int hi = (lane >> 4) << 3;
#pragma unroll
    for (int t = 0; t < 4; ++t)
#pragma unroll
        for (int r = 0; r < 8; ++r) {
            int m = t * 16 + hi + r;
            if (m < AA) {
                float v = acc[t][r] + an_bias[(((size_t)h * AA + m) << 12) + l0 + nn];
                attn[((((size_t)b * HH + h) * AA + m) << 12) + l0 + nn] = v;
            }
        }
}

// ---------------- Kernel 4: row softmax over L=4096 ----------------
__global__ __launch_bounds__(256) void k_softmax(float* __restrict__ attn) {
    __shared__ float redm[8], reds[8];
    const int tid = threadIdx.x;
    float* p = attn + (size_t)blockIdx.x * LL;
    float vals[16];
    float mx = -INFINITY;
#pragma unroll
    for (int j = 0; j < 16; ++j) { vals[j] = p[tid + (j << 8)]; mx = fmaxf(mx, vals[j]); }
#pragma unroll
    for (int o = 16; o > 0; o >>= 1) mx = fmaxf(mx, __shfl_xor(mx, o, 32));
    if ((tid & 31) == 0) redm[tid >> 5] = mx;
    __syncthreads();
    float m = redm[0];
#pragma unroll
    for (int w = 1; w < 8; ++w) m = fmaxf(m, redm[w]);
    float s = 0.f;
#pragma unroll
    for (int j = 0; j < 16; ++j) { vals[j] = __expf(vals[j] - m); s += vals[j]; }
#pragma unroll
    for (int o = 16; o > 0; o >>= 1) s += __shfl_xor(s, o, 32);
    if ((tid & 31) == 0) reds[tid >> 5] = s;
    __syncthreads();
    float tot = 0.f;
#pragma unroll
    for (int w = 0; w < 8; ++w) tot += reds[w];
    float inv = 1.f / tot;
#pragma unroll
    for (int j = 0; j < 16; ++j) p[tid + (j << 8)] = vals[j] * inv;
}

// ---------------- Kernel 5: agent_v partials (split-K over L) ----------------
// grid(B*H, NSPLIT). M = 64 (padded agents), N = 64 (hd), K = L/NSPLIT = 512.
__global__ __launch_bounds__(128) void k_agentv(const float* __restrict__ attn,
                                                const float* __restrict__ vproj,
                                                float* __restrict__ part) {
    __shared__ float At[64 * SP];  // [i][l']
    __shared__ float Vs[64 * SP];  // [l'][d]
    const int tid = threadIdx.x;
    const int lane = tid & 31, wave = tid >> 5;
    const int lane15 = lane & 15;
    const int b = blockIdx.x >> 4, h = blockIdx.x & 15;
    const int split = blockIdx.y;
    const int lbeg = split * (LL / NSPLIT);

    v8f acc[4];
#pragma unroll
    for (int t = 0; t < 4; ++t) zero8(acc[t]);
    const int khalf = (lane >> 4) << 1;
    const int nn = wave * 16 + lane15;

    for (int l0 = lbeg; l0 < lbeg + LL / NSPLIT; l0 += 64) {
        for (int idx = tid; idx < 64 * 64; idx += 128) {
            int r = idx >> 6, c = idx & 63;
            At[r * SP + c] = (r < AA)
                ? attn[((((size_t)b * HH + h) * AA + r) << 12) + l0 + c] : 0.f;
            Vs[r * SP + c] = vproj[((size_t)b * LL + l0 + r) * CC + (h << 6) + c];
        }
        __syncthreads();
        for (int ks = 0; ks < 64; ks += 4) {
            int kk = ks + khalf;
            v2f bf; bf.x = Vs[kk * SP + nn]; bf.y = Vs[(kk + 1) * SP + nn];
#pragma unroll
            for (int t = 0; t < 4; ++t) {
                int mm = t * 16 + lane15;
                v2f a; a.x = At[mm * SP + kk]; a.y = At[mm * SP + kk + 1];
                acc[t] = wmma4(a, bf, acc[t]);
            }
        }
        __syncthreads();
    }
    const int hi = (lane >> 4) << 3;
#pragma unroll
    for (int t = 0; t < 4; ++t)
#pragma unroll
        for (int r = 0; r < 8; ++r) {
            int m = t * 16 + hi + r;
            if (m < AA)
                part[(size_t)split * (BB * HH * AA * HD) +
                     ((((size_t)b * HH + h) * AA + m) << 6) + nn] = acc[t][r];
        }
}

// fixed-order split-K reduction (deterministic)
__global__ __launch_bounds__(256) void k_agentv_reduce(const float* __restrict__ part,
                                                       float* __restrict__ agentv) {
    int idx = blockIdx.x * 256 + threadIdx.x;
    if (idx >= BB * HH * AA * HD) return;
    float s = 0.f;
#pragma unroll
    for (int p = 0; p < NSPLIT; ++p)
        s += part[(size_t)p * (BB * HH * AA * HD) + idx];
    agentv[idx] = s;
}

// ---------------- Kernel 6: stage 2 + lepe ----------------
// grid(L/64, B*H). Computes s2^T (agents x queries) so softmax-over-agents is a
// per-lane reduction + one shfl_xor(16); then x^T = agent_v^T @ attn^T.
__global__ __launch_bounds__(128) void k_stage2(const float* __restrict__ Q,
                                                const float* __restrict__ agent,
                                                const float* __restrict__ agentv,
                                                const float* __restrict__ na_bias,
                                                const float* __restrict__ vproj,
                                                float* __restrict__ out) {
    __shared__ float AgAV[64 * SP];  // first scaled agent [i][d], then agent_v [i][d]
    __shared__ float Qs[64 * SP];    // [j][d]
    __shared__ float S[64 * SP];     // attn^T [i][j], then x^T [d][j]
    const int tid = threadIdx.x;
    const int lane = tid & 31, wave = tid >> 5;
    const int lane15 = lane & 15;
    const int b = blockIdx.y >> 4, h = blockIdx.y & 15;
    const int l0 = blockIdx.x * 64;
    const int khalf = (lane >> 4) << 1;
    const int hi = (lane >> 4) << 3;
    const int nn = wave * 16 + lane15;

    for (int idx = tid; idx < 64 * 64; idx += 128) {
        int r = idx >> 6, c = idx & 63;
        AgAV[r * SP + c] = (r < AA)
            ? SCALE * agent[(((size_t)b * AA + r) << 10) + (h << 6) + c] : 0.f;
        Qs[r * SP + c] = Q[((size_t)b * LL + l0 + r) * CC + (h << 6) + c];
    }
    __syncthreads();

    // GEMM1: s2T[i][j] = sum_d (scale*ag[i][d]) * q[l0+j][d]
    v8f sc[4];
#pragma unroll
    for (int t = 0; t < 4; ++t) zero8(sc[t]);
    for (int ks = 0; ks < 64; ks += 4) {
        int kk = ks + khalf;
        v2f bf; bf.x = Qs[nn * SP + kk]; bf.y = Qs[nn * SP + kk + 1];
#pragma unroll
        for (int t = 0; t < 4; ++t) {
            int mm = t * 16 + lane15;
            v2f a; a.x = AgAV[mm * SP + kk]; a.y = AgAV[mm * SP + kk + 1];
            sc[t] = wmma4(a, bf, sc[t]);
        }
    }
    // bias + mask padded agents
#pragma unroll
    for (int t = 0; t < 4; ++t)
#pragma unroll
        for (int r = 0; r < 8; ++r) {
            int m = t * 16 + hi + r;
            if (m < AA)
                sc[t][r] += na_bias[(((size_t)h * AA + m) << 12) + l0 + nn];
            else
                sc[t][r] = -INFINITY;
        }
    // softmax over agents (column j = lane pair {lane, lane^16})
    float mx = -INFINITY;
#pragma unroll
    for (int t = 0; t < 4; ++t)
#pragma unroll
        for (int r = 0; r < 8; ++r) mx = fmaxf(mx, sc[t][r]);
    mx = fmaxf(mx, __shfl_xor(mx, 16, 32));
    float sum = 0.f;
#pragma unroll
    for (int t = 0; t < 4; ++t)
#pragma unroll
        for (int r = 0; r < 8; ++r) {
            float e = __expf(sc[t][r] - mx);
            sc[t][r] = e; sum += e;
        }
    sum += __shfl_xor(sum, 16, 32);
    float inv = 1.f / sum;
    // write attn^T into LDS (each wave owns its 16 columns)
#pragma unroll
    for (int t = 0; t < 4; ++t)
#pragma unroll
        for (int r = 0; r < 8; ++r)
            S[(t * 16 + hi + r) * SP + nn] = sc[t][r] * inv;
    __syncthreads();

    // reload agent_v into AgAV (done with scaled agents)
    for (int idx = tid; idx < 64 * 64; idx += 128) {
        int r = idx >> 6, c = idx & 63;
        AgAV[r * SP + c] = (r < AA)
            ? agentv[((((size_t)b * HH + h) * AA + r) << 6) + c] : 0.f;
    }
    __syncthreads();

    // GEMM2: xT[d][j] = sum_i agent_v[i][d] * attnT[i][j]
    v8f xt[4];
#pragma unroll
    for (int t = 0; t < 4; ++t) zero8(xt[t]);
    for (int ks = 0; ks < 64; ks += 4) {
        int kk = ks + khalf;
        v2f bf; bf.x = S[kk * SP + nn]; bf.y = S[(kk + 1) * SP + nn];
#pragma unroll
        for (int t = 0; t < 4; ++t) {
            int mm = t * 16 + lane15;
            v2f a; a.x = AgAV[kk * SP + mm]; a.y = AgAV[(kk + 1) * SP + mm];
            xt[t] = wmma4(a, bf, xt[t]);
        }
    }
    // overwrite S with xT (each wave touches only its own columns)
#pragma unroll
    for (int t = 0; t < 4; ++t)
#pragma unroll
        for (int r = 0; r < 8; ++r)
            S[(t * 16 + hi + r) * SP + nn] = xt[t][r];
    __syncthreads();

    // epilogue: out = xT^T + lepe (coalesced over d)
    for (int idx = tid; idx < 64 * 64; idx += 128) {
        int j = idx >> 6, d = idx & 63;
        size_t g = ((size_t)b * LL + l0 + j) * CC + (h << 6) + d;
        out[g] = S[d * SP + j] + vproj[g];
    }
}

// ---------------- launch ----------------
extern "C" void kernel_launch(void* const* d_in, const int* in_sizes, int n_in,
                              void* d_out, int out_size, void* d_ws, size_t ws_size,
                              hipStream_t stream) {
    const float* q  = (const float*)d_in[0];
    const float* k  = (const float*)d_in[1];
    const float* v  = (const float*)d_in[2];
    const float* W  = (const float*)d_in[3];
    const float* bv = (const float*)d_in[4];
    const float* an = (const float*)d_in[5];
    const float* na = (const float*)d_in[6];
    float* out = (float*)d_out;

    float* ws      = (float*)d_ws;
    float* vproj   = ws;                                  // 16,777,216
    float* agent   = vproj + (size_t)BB * LL * CC;        //    200,704
    float* attn    = agent + (size_t)BB * AA * CC;        // 12,845,056
    float* avpart  = attn + (size_t)BB * HH * AA * LL;    //  1,605,632
    float* agentv  = avpart + (size_t)NSPLIT * BB * HH * AA * HD; // 200,704

    k_vproj  <<<dim3((BB * LL) / 128, CC / 128), 256, 0, stream>>>(v, W, bv, vproj);
    k_pool   <<<(BB * AA * CC + 255) / 256, 256, 0, stream>>>(q, agent);
    k_s1     <<<dim3(LL / 64, BB * HH), 128, 0, stream>>>(k, agent, an, attn);
    k_softmax<<<BB * HH * AA, 256, 0, stream>>>(attn);
    k_agentv <<<dim3(BB * HH, NSPLIT), 128, 0, stream>>>(attn, vproj, avpart);
    k_agentv_reduce<<<(BB * HH * AA * HD + 255) / 256, 256, 0, stream>>>(avpart, agentv);
    k_stage2 <<<dim3(LL / 64, BB * HH), 128, 0, stream>>>(q, agent, agentv, na, vproj, out);
}
